// DubinsLSTMEncoderDecoder_61409442398233
// MI455X (gfx1250) — matrix operand order repaired and checked by reading, not compiled, not run on hardware
//
#include <hip/hip_runtime.h>
#include <hip/hip_bf16.h>

typedef _Float16 half_t;
typedef __attribute__((ext_vector_type(16))) _Float16 v16h;
typedef __attribute__((ext_vector_type(8)))  float    v8f;

#define WAVES 8          // waves per block (2 per SIMD32)
#define ROWS  16         // batch rows per wave (one WMMA M tile)

// ---- fast activations: native v_tanh_f32 (confirmed present on gfx1250) ----
#if __has_builtin(__builtin_amdgcn_tanhf)
  #define FAST_TANH(x) __builtin_amdgcn_tanhf(x)
  #define HAVE_NATIVE_TANH 1
#elif __has_builtin(__builtin_amdgcn_tanh_f32)
  #define FAST_TANH(x) __builtin_amdgcn_tanh_f32(x)
  #define HAVE_NATIVE_TANH 1
#else
  #define HAVE_NATIVE_TANH 0
#endif

__device__ __forceinline__ float fsig_raw(float x) {
    float e = __expf(-x);
    return __builtin_amdgcn_rcpf(1.0f + e);
}
__device__ __forceinline__ float ftanh_raw(float x) {
    float e = __expf(-2.0f * x);
    return 1.0f - 2.0f * __builtin_amdgcn_rcpf(1.0f + e);
}
// sigmoid(x + b) with hb = 0.5*b pre-scaled at staging time
__device__ __forceinline__ float sig_biash(float x, float hb) {
#if HAVE_NATIVE_TANH
    return fmaf(FAST_TANH(fmaf(x, 0.5f, hb)), 0.5f, 0.5f);
#else
    return fsig_raw(fmaf(2.0f, hb, x));
#endif
}
__device__ __forceinline__ float tanh_bias(float x, float b) {
#if HAVE_NATIVE_TANH
    return FAST_TANH(x + b);
#else
    return ftanh_raw(x + b);
#endif
}
__device__ __forceinline__ float tanh1(float x) {
#if HAVE_NATIVE_TANH
    return FAST_TANH(x);
#else
    return ftanh_raw(x);
#endif
}

__device__ __forceinline__ v16h loadFrag(const half_t* base, int lane) {
    return *reinterpret_cast<const v16h*>(base + lane * 16);
}

__device__ __forceinline__ v8f wmma16(v16h a, v16h b, v8f c) {
    return __builtin_amdgcn_wmma_f32_16x16x32_f16(
        false, a, false, b, (short)0, c, false, false);
}

// Raw gate tile (no bias): acc = x@Wih_tile + h@Whh_tile, from inline-0 C
template <int LAYER>
__device__ __forceinline__ v8f gateTile(int gt, const v16h* xin, const v16h* hA,
                                        int lane,
                                        const half_t* sIH0, const half_t* sIH,
                                        const half_t* sHH) {
    v8f acc = {};
    if (LAYER == 0) {
        v16h b = loadFrag(sIH0 + gt * 512, lane);
        acc = wmma16(xin[0], b, acc);        // K=3 padded to 32
    } else {
#pragma unroll
        for (int kc = 0; kc < 2; ++kc) {
            v16h b = loadFrag(sIH + (((LAYER - 1) * 16 + gt) * 2 + kc) * 512, lane);
            acc = wmma16(xin[kc], b, acc);
        }
    }
#pragma unroll
    for (int kc = 0; kc < 2; ++kc) {
        v16h b = loadFrag(sHH + ((LAYER * 16 + gt) * 2 + kc) * 512, lane);
        acc = wmma16(hA[kc], b, acc);
    }
    return acc;
}

__global__ __launch_bounds__(WAVES * 32, 1)
void lstm_decode_kernel(const float* __restrict__ conds,
                        const float* __restrict__ W1, const float* __restrict__ b1,
                        const float* __restrict__ W2, const float* __restrict__ b2,
                        const float* __restrict__ Wih0,
                        const float* __restrict__ WihR,
                        const float* __restrict__ Whh,
                        const float* __restrict__ bih, const float* __restrict__ bhh,
                        const float* __restrict__ Wo,  const float* __restrict__ bo,
                        float* __restrict__ out, int Btot, int T) {
    // ---- LDS: weight B-fragments + bias + per-wave f16 scratch (~287 KB) ----
    __shared__ __align__(32) half_t sIH0[16 * 512];          //  16 KB
    __shared__ __align__(32) half_t sIH [3 * 16 * 2 * 512];  //  96 KB
    __shared__ __align__(32) half_t sHH [4 * 16 * 2 * 512];  // 128 KB
    __shared__ __align__(32) half_t sWoF[2 * 512];           //   2 KB (Wo^T, 3 live cols)
    __shared__ float sBias[4 * 256];                         //   4 KB (pre-scaled: i/f/o x0.5)
    __shared__ float sBo[3];
    __shared__ __align__(32) half_t sWbuf[WAVES][16 * 128];  //  32 KB (embed / h frags / pred)
    __shared__ float sHid[WAVES][64];                        //   2 KB

    const int tid  = threadIdx.x;
    const int nthr = blockDim.x;
    const int lane = tid & 31;
    const int w    = tid >> 5;

    // ---- Stage weights into B-fragment layout ----
    // lane L holds column N=L%16; halfs t=0..15 hold K = (L<16?0:16)+t of the 32-chunk
    for (int idx = tid; idx < 4 * 16 * 2 * 512; idx += nthr) {
        int t = idx & 15, ln = (idx >> 4) & 31, kc = (idx >> 9) & 1;
        int tile = (idx >> 10) & 15, l = idx >> 14;
        int col = tile * 16 + (ln & 15);
        int k   = kc * 32 + ((ln < 16) ? 0 : 16) + t;
        sHH[idx] = (half_t)Whh[(l * 256 + col) * 64 + k];
    }
    for (int idx = tid; idx < 3 * 16 * 2 * 512; idx += nthr) {
        int t = idx & 15, ln = (idx >> 4) & 31, kc = (idx >> 9) & 1;
        int tile = (idx >> 10) & 15, l = idx >> 14;
        int col = tile * 16 + (ln & 15);
        int k   = kc * 32 + ((ln < 16) ? 0 : 16) + t;
        sIH[idx] = (half_t)WihR[(l * 256 + col) * 64 + k];
    }
    for (int idx = tid; idx < 16 * 512; idx += nthr) {
        int t = idx & 15, ln = (idx >> 4) & 31, tile = idx >> 9;
        int col = tile * 16 + (ln & 15);
        int k   = ((ln < 16) ? 0 : 16) + t;
        sIH0[idx] = (k < 3) ? (half_t)Wih0[col * 3 + k] : (half_t)0.0f;
    }
    for (int idx = tid; idx < 2 * 512; idx += nthr) {
        int t = idx & 15, ln = (idx >> 4) & 31, kc = idx >> 9;
        int ncol = ln & 15;
        int k    = kc * 32 + ((ln < 16) ? 0 : 16) + t;
        sWoF[idx] = (ncol < 3) ? (half_t)Wo[ncol * 64 + k] : (half_t)0.0f;
    }
    for (int idx = tid; idx < 1024; idx += nthr) {
        // gate rows per layer: i=0..63, f=64..127, g=128..191, o=192..255
        float scale = (((idx & 255) >> 6) == 2) ? 1.0f : 0.5f;   // g keeps full bias
        sBias[idx] = (bih[idx] + bhh[idx]) * scale;
    }
    for (int idx = tid; idx < 3; idx += nthr) sBo[idx] = bo[idx];
    __syncthreads();

    const int batch0 = (blockIdx.x * WAVES + w) * ROWS;
    if (batch0 >= Btot) return;

    half_t* embed = &sWbuf[w][0];     // [16][128] f16 during init
    half_t* hbuf  = &sWbuf[w][0];     // 2 frags x 512 halfs during main loop (A-frag layout)
    half_t* pbuf  = &sWbuf[w][1024];  // 16 rows x 3 pred halfs
    float*  hid   = &sHid[w][0];

    // ---- Init: cond_embed = relu(conds@W1^T+b1)@W2^T + b2  (stored f16) ----
    for (int r = 0; r < ROWS; ++r) {
        int row = batch0 + r;
#pragma unroll
        for (int rep = 0; rep < 2; ++rep) {
            int j = lane + rep * 32;
            float a = b1[j];
#pragma unroll
            for (int k = 0; k < 8; ++k) a += conds[row * 8 + k] * W1[j * 8 + k];
            hid[j] = fmaxf(a, 0.0f);
        }
#pragma unroll
        for (int rep = 0; rep < 4; ++rep) {
            int e = lane + rep * 32;
            float a = b2[e];
            for (int k = 0; k < 64; ++k) a += hid[k] * W2[e * 64 + k];
            embed[r * 128 + e] = (half_t)a;
        }
    }

    const int q   = lane & 15;
    const int hi8 = (lane < 16) ? 0 : 8;

    // ---- c state: f32 C-layout tiles ----
    v8f ctile[4];
#pragma unroll
    for (int n = 0; n < 4; ++n) {
#pragma unroll
        for (int e = 0; e < 8; ++e)
            ctile[n][e] = (float)embed[(e + hi8) * 128 + 64 + n * 16 + q];
    }
    v8f c[4][4];
#pragma unroll
    for (int l = 0; l < 4; ++l)
#pragma unroll
        for (int n = 0; n < 4; ++n) c[l][n] = ctile[n];

    // ---- h state: f16 A-fragments ----
    v16h h0frag[2];
#pragma unroll
    for (int f = 0; f < 2; ++f) {
        v16h v;
#pragma unroll
        for (int t = 0; t < 16; ++t) {
            int K = f * 32 + t + ((lane < 16) ? 0 : 16);
            v[t] = embed[(lane & 15) * 128 + K];
        }
        h0frag[f] = v;
    }
    v16h hA[4][2];
#pragma unroll
    for (int l = 0; l < 4; ++l) { hA[l][0] = h0frag[0]; hA[l][1] = h0frag[1]; }

    // ---- x0 A-fragment: persistent; only halfs 0..2 rewritten per step ----
    v16h x0;
#pragma unroll
    for (int t = 0; t < 16; ++t) x0[t] = (half_t)0.0f;
    if (lane < 16) { x0[0] = (half_t)0.5f; x0[1] = (half_t)0.5f; }

    // Base of this lane's h-store slots: slot (La = M + ((n&1)<<4), t = q)
    const int hstoreBase = (hi8 << 4) + q;

    // ---- Time loop ----
    for (int t = 0; t < T; ++t) {
#pragma unroll
        for (int l = 0; l < 4; ++l) {
            const v16h* xin = (l == 0) ? &x0 : hA[l - 1];
#pragma unroll
            for (int n = 0; n < 4; ++n) {
                float hbi = sBias[l * 256 + n * 16 + q];          // 0.5*b_i
                float hbf = sBias[l * 256 + 64 + n * 16 + q];     // 0.5*b_f
                float bg  = sBias[l * 256 + 128 + n * 16 + q];    //     b_g
                float hbo = sBias[l * 256 + 192 + n * 16 + q];    // 0.5*b_o
                v8f gi, gf, gg, go;
                switch (l) {
                case 0:
                    gi = gateTile<0>(n,      xin, hA[0], lane, sIH0, sIH, sHH);
                    gf = gateTile<0>(n + 4,  xin, hA[0], lane, sIH0, sIH, sHH);
                    gg = gateTile<0>(n + 8,  xin, hA[0], lane, sIH0, sIH, sHH);
                    go = gateTile<0>(n + 12, xin, hA[0], lane, sIH0, sIH, sHH);
                    break;
                case 1:
                    gi = gateTile<1>(n,      xin, hA[1], lane, sIH0, sIH, sHH);
                    gf = gateTile<1>(n + 4,  xin, hA[1], lane, sIH0, sIH, sHH);
                    gg = gateTile<1>(n + 8,  xin, hA[1], lane, sIH0, sIH, sHH);
                    go = gateTile<1>(n + 12, xin, hA[1], lane, sIH0, sIH, sHH);
                    break;
                case 2:
                    gi = gateTile<2>(n,      xin, hA[2], lane, sIH0, sIH, sHH);
                    gf = gateTile<2>(n + 4,  xin, hA[2], lane, sIH0, sIH, sHH);
                    gg = gateTile<2>(n + 8,  xin, hA[2], lane, sIH0, sIH, sHH);
                    go = gateTile<2>(n + 12, xin, hA[2], lane, sIH0, sIH, sHH);
                    break;
                default:
                    gi = gateTile<3>(n,      xin, hA[3], lane, sIH0, sIH, sHH);
                    gf = gateTile<3>(n + 4,  xin, hA[3], lane, sIH0, sIH, sHH);
                    gg = gateTile<3>(n + 8,  xin, hA[3], lane, sIH0, sIH, sHH);
                    go = gateTile<3>(n + 12, xin, hA[3], lane, sIH0, sIH, sHH);
                    break;
                }
                // h store goes straight into A-fragment layout
                int sbase = ((n >> 1) << 9) + hstoreBase + ((n & 1) << 8);
                v8f cc = c[l][n];
#pragma unroll
                for (int e = 0; e < 8; ++e) {
                    float sf = sig_biash(gf[e], hbf);
                    float si = sig_biash(gi[e], hbi);
                    float tg = tanh_bias(gg[e], bg);
                    float cn = fmaf(sf, cc[e], si * tg);
                    float so = sig_biash(go[e], hbo);
                    float hn = so * tanh1(cn);
                    cc[e] = cn;
                    hbuf[sbase + e * 16] = (half_t)hn;
                }
                c[l][n] = cc;
            }
            // reload h[l] as A-fragments: direct contiguous v16h per lane
#pragma unroll
            for (int f = 0; f < 2; ++f)
                hA[l][f] = *reinterpret_cast<const v16h*>(&hbuf[f * 512 + lane * 16]);
        }

        // ---- pred = h3 @ Wo^T + bo via WMMA (C layout: lanes q<3 hold all rows) ----
        v8f pc = {};
        pc = wmma16(hA[3][0], loadFrag(sWoF, lane), pc);
        pc = wmma16(hA[3][1], loadFrag(sWoF + 512, lane), pc);
        if (q < 3) {
            float bq = sBo[q];
#pragma unroll
            for (int e = 0; e < 8; ++e) {
                float p = pc[e] + bq;
                int row = e + hi8;
                out[((size_t)(batch0 + row) * (size_t)T + (size_t)t) * 3 + q] = p;
                pbuf[row * 3 + q] = (half_t)p;
            }
        }
        // rebuild x0 (K=3) from pred bounce: d16 loads write register halves directly
        if (lane < 16) {
            x0[0] = pbuf[lane * 3 + 0];
            x0[1] = pbuf[lane * 3 + 1];
            x0[2] = pbuf[lane * 3 + 2];
        }
    }
}

extern "C" void kernel_launch(void* const* d_in, const int* in_sizes, int n_in,
                              void* d_out, int out_size, void* d_ws, size_t ws_size,
                              hipStream_t stream) {
    const float* conds = (const float*)d_in[0];
    const float* W1    = (const float*)d_in[1];
    const float* b1    = (const float*)d_in[2];
    const float* W2    = (const float*)d_in[3];
    const float* b2    = (const float*)d_in[4];
    const float* Wih0  = (const float*)d_in[5];
    const float* WihR  = (const float*)d_in[6];
    const float* Whh   = (const float*)d_in[7];
    const float* bih   = (const float*)d_in[8];
    const float* bhh   = (const float*)d_in[9];
    const float* Wo    = (const float*)d_in[10];
    const float* bo    = (const float*)d_in[11];

    int B = in_sizes[0] / 8;
    int T = out_size / (B * 3);

    int blocks = (B + (WAVES * ROWS) - 1) / (WAVES * ROWS);
    lstm_decode_kernel<<<blocks, WAVES * 32, 0, stream>>>(
        conds, W1, b1, W2, b2, Wih0, WihR, Whh, bih, bhh, Wo, bo,
        (float*)d_out, B, T);
}
